// GCN_31018253812316
// MI455X (gfx1250) — compile-verified, matrix operand order checked by reading, not run on hardware
//
#include <hip/hip_runtime.h>
#include <hip/hip_fp16.h>

typedef __attribute__((ext_vector_type(16))) _Float16 v16h;
typedef __attribute__((ext_vector_type(8)))  float    v8f;
typedef __attribute__((ext_vector_type(2)))  float    v2f;

#define THREADS 256
#define WAVES_PER_BLOCK (THREADS / 32)

// ---------------------------------------------------------------------------
// Degree / normalization
// ---------------------------------------------------------------------------
__global__ __launch_bounds__(THREADS)
void k_fill_ones(float* __restrict__ p, int n) {
  int i = blockIdx.x * blockDim.x + threadIdx.x;
  if (i < n) p[i] = 1.0f;                 // deg starts at 1 (self-loop)
}

__global__ __launch_bounds__(THREADS)
void k_deg_count(const int* __restrict__ dst, float* __restrict__ deg, int e) {
  int i = blockIdx.x * blockDim.x + threadIdx.x;
  if (i < e) atomicAdd(&deg[dst[i]], 1.0f);   // global_atomic_add_f32, L2-resident
}

__global__ __launch_bounds__(THREADS)
void k_rsqrt(float* __restrict__ p, int n) {
  int i = blockIdx.x * blockDim.x + threadIdx.x;
  if (i < n) p[i] = rsqrtf(p[i]);         // dinv = deg^-1/2
}

// ---------------------------------------------------------------------------
// Layer-1 transform: m[N,4] = x[N,32] @ W1[32,4]
// One V_WMMA_F32_16X16X32_F16 per 16-node tile (K=32 exact fit).
// A layout (ISA 7.12.2, 16-bit A 16x32): lane = (M%16) + 16*hi;
//   halves j=0..7 -> K = 8*hi + j ; j=8..15 -> K = 16 + 8*hi + (j-8)
//   => per lane, A is two contiguous 8-float runs -> 4x global_load_b128.
// C/D layout: M = vgpr + 8*hi, N = lane%16.
// D is staged in a per-wave LDS slab, then stored one float4 row per lane.
// ---------------------------------------------------------------------------
__global__ __launch_bounds__(THREADS)
void k_xform32_wmma(const float* __restrict__ x, const float* __restrict__ W,
                    float* __restrict__ m, int n) {
  __shared__ float lds[WAVES_PER_BLOCK * 64];
  int wave  = (int)((blockIdx.x * (unsigned)blockDim.x + threadIdx.x) >> 5);
  int tl    = threadIdx.x & 31;
  int row0  = wave * 16;
  if (row0 >= n) return;
  int lane16 = tl & 15;
  int hi     = tl >> 4;

  // ---- B: W1[32,4] -> 32x16 f16, cols 4..15 zero (unconditional load + select)
  v16h b;
  {
    int  colW = lane16 & 3;
    bool keep = lane16 < 4;
#pragma unroll
    for (int j = 0; j < 16; ++j) {
      int k = (j < 8) ? (hi * 8 + j) : (16 + hi * 8 + (j - 8));
      float wv = W[k * 4 + colW];
      b[j] = keep ? (_Float16)wv : (_Float16)0.0f;
    }
  }

  // ---- A: clamp row, bulk-load 2x(2x float4), select-zero out-of-range rows
  int  arow = row0 + lane16;
  bool rok  = arow < n;
  int  ar   = rok ? arow : (n - 1);           // always a legal address
  const float* xr = x + (size_t)ar * 32;
  float4 q0 = *reinterpret_cast<const float4*>(xr + hi * 8);       // K = 8hi..8hi+3
  float4 q1 = *reinterpret_cast<const float4*>(xr + hi * 8 + 4);   // K = 8hi+4..8hi+7
  float4 q2 = *reinterpret_cast<const float4*>(xr + 16 + hi * 8);  // K = 16+8hi..
  float4 q3 = *reinterpret_cast<const float4*>(xr + 16 + hi * 8 + 4);
  float zf = rok ? 1.0f : 0.0f;               // zero masked rows via multiply
  v16h a;
  a[0]=(_Float16)(q0.x*zf); a[1]=(_Float16)(q0.y*zf); a[2]=(_Float16)(q0.z*zf); a[3]=(_Float16)(q0.w*zf);
  a[4]=(_Float16)(q1.x*zf); a[5]=(_Float16)(q1.y*zf); a[6]=(_Float16)(q1.z*zf); a[7]=(_Float16)(q1.w*zf);
  a[8]=(_Float16)(q2.x*zf); a[9]=(_Float16)(q2.y*zf); a[10]=(_Float16)(q2.z*zf); a[11]=(_Float16)(q2.w*zf);
  a[12]=(_Float16)(q3.x*zf); a[13]=(_Float16)(q3.y*zf); a[14]=(_Float16)(q3.z*zf); a[15]=(_Float16)(q3.w*zf);

  v8f c = {};
  c = __builtin_amdgcn_wmma_f32_16x16x32_f16(false, a, false, b,
                                             (short)0, c, false, false);

  // ---- D: stage in LDS (per-wave slab, wave-synchronous), store coalesced rows
  float* slab = lds + ((threadIdx.x >> 5) << 6);
  if (lane16 < 4) {
#pragma unroll
    for (int r = 0; r < 8; ++r) slab[(hi * 8 + r) * 4 + lane16] = c[r];
  }
  asm volatile("s_wait_dscnt 0" ::: "memory");  // DS in-order within wave; be explicit
  if (hi == 0) {
    int mr = row0 + lane16;
    if (mr < n) {
      float4 v = *reinterpret_cast<const float4*>(slab + lane16 * 4);
      *reinterpret_cast<float4*>(m + (size_t)mr * 4) = v;
    }
  }
}

// ---------------------------------------------------------------------------
// Small transform: mout[N,FOUT] = hin[N,KIN] @ W[KIN,FOUT] (+ bias)
// V_WMMA_F32_16X16X4_F32 per 16-node tile; K padded to 4 with zeros.
// A layout (32-bit A 16x4): a[j] holds K = 2*hi + j for row lane%16.
// Unconditional clamped loads + selects; D staged via LDS as above.
// ---------------------------------------------------------------------------
template<int KIN, int FOUT, bool BIAS>
__global__ __launch_bounds__(THREADS)
void k_xform_small(const float* __restrict__ hin, const float* __restrict__ W,
                   const float* __restrict__ bias, float* __restrict__ mout, int n) {
  __shared__ float lds[WAVES_PER_BLOCK * 64];
  int wave  = (int)((blockIdx.x * (unsigned)blockDim.x + threadIdx.x) >> 5);
  int tl    = threadIdx.x & 31;
  int row0  = wave * 16;
  if (row0 >= n) return;
  int lane16 = tl & 15;
  int hi     = tl >> 4;

#if __has_builtin(__builtin_amdgcn_wmma_f32_16x16x4_f32)
  int  arow = row0 + lane16;
  bool rok  = arow < n;
  int  ar   = rok ? arow : (n - 1);
  float zf  = rok ? 1.0f : 0.0f;

  v2f a;
  if (KIN == 4) {
    float2 v = *reinterpret_cast<const float2*>(hin + (size_t)ar * 4 + hi * 2);
    a[0] = v.x * zf;  a[1] = v.y * zf;
  } else {            // KIN == 2: only K=0,1 (hi==0) are real; hi==1 lanes are zero
    float2 v = *reinterpret_cast<const float2*>(hin + (size_t)ar * 2);
    float zk = (hi == 0) ? zf : 0.0f;
    a[0] = v.x * zk;  a[1] = v.y * zk;
  }

  v2f b;
#pragma unroll
  for (int j = 0; j < 2; ++j) {
    int  k    = hi * 2 + j;
    int  kc   = (k < KIN) ? k : (KIN - 1);
    int  cc   = (lane16 < FOUT) ? lane16 : (FOUT - 1);
    float wv  = W[kc * FOUT + cc];
    b[j] = (k < KIN && lane16 < FOUT) ? wv : 0.0f;
  }

  v8f c = {};
  c = __builtin_amdgcn_wmma_f32_16x16x4_f32(false, a, false, b,
                                            (short)0, c, false, false);

  float* slab = lds + ((threadIdx.x >> 5) << 6);
  if (lane16 < FOUT) {
#pragma unroll
    for (int r = 0; r < 8; ++r) slab[(hi * 8 + r) * FOUT + lane16] = c[r];
  }
  asm volatile("s_wait_dscnt 0" ::: "memory");
  if (hi == 0) {
    int mr = row0 + lane16;
    if (mr < n) {
      if (FOUT == 4) {
        float4 v = *reinterpret_cast<const float4*>(slab + lane16 * 4);
        if (BIAS) { v.x += bias[0]; v.y += bias[1]; v.z += bias[2]; v.w += bias[3]; }
        *reinterpret_cast<float4*>(mout + (size_t)mr * 4) = v;
      } else {    // FOUT == 2
        float2 v = *reinterpret_cast<const float2*>(slab + lane16 * 2);
        if (BIAS) { v.x += bias[0]; v.y += bias[1]; }
        *reinterpret_cast<float2*>(mout + (size_t)mr * 2) = v;
      }
    }
  }
#else
  if (hi == 0) {                 // scalar fallback: lanes 0..15 -> 16 rows
    int mr = row0 + lane16;
    if (mr < n) {
#pragma unroll
      for (int f = 0; f < FOUT; ++f) {
        float acc = BIAS ? bias[f] : 0.0f;
#pragma unroll
        for (int k = 0; k < KIN; ++k) acc += hin[mr * KIN + k] * W[k * FOUT + f];
        mout[mr * FOUT + f] = acc;
      }
    }
  }
#endif
}

// ---------------------------------------------------------------------------
// agg[i,f] = dinv[i]^2 * m[i,f] + b[f]   (self-loop term + bias)
// ---------------------------------------------------------------------------
template<int F>
__global__ __launch_bounds__(THREADS)
void k_agg_init(float* __restrict__ agg, const float* __restrict__ m,
                const float* __restrict__ dinv, const float* __restrict__ b, int n) {
  int i = blockIdx.x * blockDim.x + threadIdx.x;
  if (i >= n * F) return;
  int node = i / F;
  int f    = i - node * F;
  float di = dinv[node];
  agg[i] = di * di * m[i] + b[f];
}

// ---------------------------------------------------------------------------
// Edge scatter: agg[dst] += dinv[src]*dinv[dst] * m[src]   (L2-resident atomics)
// ---------------------------------------------------------------------------
template<int F>
__global__ __launch_bounds__(THREADS)
void k_edge_scatter(const int* __restrict__ src, const int* __restrict__ dst,
                    const float* __restrict__ dinv, const float* __restrict__ m,
                    float* __restrict__ agg, int e) {
  int i = blockIdx.x * blockDim.x + threadIdx.x;
  if (i >= e) return;
  int s = src[i], d = dst[i];
  float w = dinv[s] * dinv[d];
  if (F == 4) {
    float4 v = *reinterpret_cast<const float4*>(m + (size_t)s * 4);
    atomicAdd(&agg[(size_t)d * 4 + 0], w * v.x);
    atomicAdd(&agg[(size_t)d * 4 + 1], w * v.y);
    atomicAdd(&agg[(size_t)d * 4 + 2], w * v.z);
    atomicAdd(&agg[(size_t)d * 4 + 3], w * v.w);
  } else {
    float2 v = *reinterpret_cast<const float2*>(m + (size_t)s * 2);
    atomicAdd(&agg[(size_t)d * 2 + 0], w * v.x);
    atomicAdd(&agg[(size_t)d * 2 + 1], w * v.y);
  }
}

__global__ __launch_bounds__(THREADS)
void k_tanh(float* __restrict__ dst, const float* __restrict__ srcp, int cnt) {
  int i = blockIdx.x * blockDim.x + threadIdx.x;
  if (i < cnt) dst[i] = tanhf(srcp[i]);
}

// ---------------------------------------------------------------------------
// Host-side orchestration (graph-capture safe: only kernel launches on stream)
// ---------------------------------------------------------------------------
extern "C" void kernel_launch(void* const* d_in, const int* in_sizes, int n_in,
                              void* d_out, int out_size, void* d_ws, size_t ws_size,
                              hipStream_t stream) {
  const float* x  = (const float*)d_in[0];
  const int*   ei = (const int*)  d_in[1];
  const float* W1 = (const float*)d_in[2];
  const float* b1 = (const float*)d_in[3];
  const float* W2 = (const float*)d_in[4];
  const float* b2 = (const float*)d_in[5];
  const float* W3 = (const float*)d_in[6];
  const float* b3 = (const float*)d_in[7];
  const float* Wc = (const float*)d_in[8];
  const float* bc = (const float*)d_in[9];

  const int N = in_sizes[0] / 32;
  const int E = in_sizes[1] / 2;
  const int* src = ei;
  const int* dst = ei + E;

  float* out = (float*)d_out;                 // [N,4]
  float* h3  = out + (size_t)N * 4;           // [N,2] second tuple element

  float* ws   = (float*)d_ws;                 // 13*N floats ~ 5.2 MB
  float* dinv = ws;                           // N
  float* m    = ws + (size_t)N;               // N*4
  float* agg  = ws + (size_t)5 * N;           // N*4
  float* h    = ws + (size_t)9 * N;           // N*4

  const int gN  = (N + THREADS - 1) / THREADS;
  const int gE  = (E + THREADS - 1) / THREADS;
  const int g4N = (N * 4 + THREADS - 1) / THREADS;
  const int g2N = (N * 2 + THREADS - 1) / THREADS;
  const int tiles = (N + 15) / 16;
  const int gW  = (tiles + WAVES_PER_BLOCK - 1) / WAVES_PER_BLOCK;

  // degree -> dinv
  k_fill_ones<<<gN, THREADS, 0, stream>>>(dinv, N);
  k_deg_count<<<gE, THREADS, 0, stream>>>(dst, dinv, E);
  k_rsqrt<<<gN, THREADS, 0, stream>>>(dinv, N);

  // layer 1: 32 -> 4
  k_xform32_wmma<<<gW, THREADS, 0, stream>>>(x, W1, m, N);
  k_agg_init<4><<<g4N, THREADS, 0, stream>>>(agg, m, dinv, b1, N);
  k_edge_scatter<4><<<gE, THREADS, 0, stream>>>(src, dst, dinv, m, agg, E);
  k_tanh<<<g4N, THREADS, 0, stream>>>(h, agg, N * 4);

  // layer 2: 4 -> 4
  k_xform_small<4, 4, false><<<gW, THREADS, 0, stream>>>(h, W2, nullptr, m, N);
  k_agg_init<4><<<g4N, THREADS, 0, stream>>>(agg, m, dinv, b2, N);
  k_edge_scatter<4><<<gE, THREADS, 0, stream>>>(src, dst, dinv, m, agg, E);
  k_tanh<<<g4N, THREADS, 0, stream>>>(h, agg, N * 4);

  // layer 3: 4 -> 2  (tanh result is the second output, written in place)
  k_xform_small<4, 2, false><<<gW, THREADS, 0, stream>>>(h, W3, nullptr, m, N);
  k_agg_init<2><<<g2N, THREADS, 0, stream>>>(agg, m, dinv, b3, N);
  k_edge_scatter<2><<<gE, THREADS, 0, stream>>>(src, dst, dinv, m, agg, E);
  k_tanh<<<g2N, THREADS, 0, stream>>>(h3, agg, N * 2);

  // classifier: out = h3 @ Wc + bc
  k_xform_small<2, 4, true><<<gW, THREADS, 0, stream>>>(h3, Wc, bc, out, N);
}
